// GNN_12850542150610
// MI455X (gfx1250) — matrix operand order repaired and checked by reading, not compiled, not run on hardware
//
#include <hip/hip_runtime.h>

#define KNN 20
#define P_CLOUD 1024
#define NCLOUD 16

typedef __attribute__((ext_vector_type(16))) __bf16 bf16x16;
typedef __attribute__((ext_vector_type(8)))  __bf16 bf16x8;
typedef __attribute__((ext_vector_type(2)))  __bf16 bf16x2;
typedef __attribute__((ext_vector_type(8)))  float  f32x8;
typedef __attribute__((ext_vector_type(4)))  int    i32x4;

#define AS1 __attribute__((address_space(1)))
#define AS3 __attribute__((address_space(3)))

__device__ __forceinline__ __bf16 f2bf(float f) {
#if __has_builtin(__builtin_amdgcn_cvt_pk_bf16_f32)
  bf16x2 p = __builtin_amdgcn_cvt_pk_bf16_f32(f, f);   // v_cvt_pk_bf16_f32
  return p[0];
#else
  unsigned u = __builtin_bit_cast(unsigned, f);
  unsigned r = u + 0x7FFFu + ((u >> 16) & 1u);         // round-to-nearest-even
  unsigned short s = (unsigned short)(r >> 16);
  return __builtin_bit_cast(__bf16, s);
#endif
}

__device__ __forceinline__ void wait_async0() {
#if __has_builtin(__builtin_amdgcn_s_wait_asynccnt)
  __builtin_amdgcn_s_wait_asynccnt(0);
#else
  asm volatile("s_wait_asynccnt 0" ::: "memory");
#endif
}

// ---- WMMA fragment loaders (ISA 7.12.2, 16-bit operand layouts, wave32) ----
// A 16x32 (MxK): lanes 0-15 -> M=lane, K=0..7 & 16..23 ; lanes 16-31 -> K=8..15 & 24..31
__device__ __forceinline__ bf16x16 load_a_frag(const __bf16* __restrict__ base,
                                               int ld, int m0, int k0, int lane) {
  const int m  = m0 + (lane & 15);
  const int kb = k0 + ((lane & 16) ? 8 : 0);
  const __bf16* p = base + m * ld + kb;
  bf16x8 lo = *(const bf16x8*)(p);
  bf16x8 hi = *(const bf16x8*)(p + 16);
  bf16x16 r;
#pragma unroll
  for (int i = 0; i < 8; ++i) { r[i] = lo[i]; r[i + 8] = hi[i]; }
  return r;
}
// B 32x16 (KxN): lanes 0-15 -> N=lane, K=0..7 & 16..23 ; lanes 16-31 -> K=8..15 & 24..31.
// Weights are pre-transposed [N][K], so each lane reads two 8-wide K runs.
__device__ __forceinline__ bf16x16 load_b_frag(const __bf16* __restrict__ wt,
                                               int ld, int n0, int k0, int lane) {
  const int n  = n0 + (lane & 15);
  const int kb = k0 + ((lane & 16) ? 8 : 0);
  const __bf16* p = wt + n * ld + kb;
  bf16x8 lo = *(const bf16x8*)(p);
  bf16x8 hi = *(const bf16x8*)(p + 16);
  bf16x16 r;
#pragma unroll
  for (int i = 0; i < 8; ++i) { r[i] = lo[i]; r[i + 8] = hi[i]; }
  return r;
}

// One dense layer on this wave's 16-row stripe: dst = act(src @ W^T + b), bf16 out to LDS.
__device__ __forceinline__ void mlp_layer(const __bf16* __restrict__ src, int ld_src, int Ksrc,
                                          const __bf16* __restrict__ wt, const float* __restrict__ bias,
                                          __bf16* __restrict__ dst, int ld_dst, int Nout,
                                          bool relu, int m0, int lane) {
  for (int n0 = 0; n0 < Nout; n0 += 16) {
    const int n = n0 + (lane & 15);
    const float bv = bias[n];
    f32x8 acc;
#pragma unroll
    for (int i = 0; i < 8; ++i) acc[i] = bv;
    for (int k0 = 0; k0 < Ksrc; k0 += 32) {
      bf16x16 a = load_a_frag(src, ld_src, m0, k0, lane);
      bf16x16 b = load_b_frag(wt, Ksrc, n0, k0, lane);
      acc = __builtin_amdgcn_wmma_f32_16x16x32_bf16(false, a, false, b,
                                                    (short)0, acc, false, false);
    }
    // D layout: VGPR r -> M = r (+8 for lanes 16-31), N = lane&15
#pragma unroll
    for (int r = 0; r < 8; ++r) {
      float v = acc[r];
      if (relu) v = fmaxf(v, 0.0f);
      const int m = m0 + r + ((lane & 16) ? 8 : 0);
      dst[m * ld_dst + n] = f2bf(v);
    }
  }
}

// Final layer: f32 out to global, only first KOUT columns valid.
template <int KOUT>
__device__ __forceinline__ void mlp_layer_out(const __bf16* __restrict__ src, int ld_src, int Ksrc,
                                              const __bf16* __restrict__ wt, const float* __restrict__ bias,
                                              float* __restrict__ out, int m0, int lane) {
  const int n = lane & 15;
  const float bv = (n < KOUT) ? bias[n] : 0.0f;
  f32x8 acc;
#pragma unroll
  for (int i = 0; i < 8; ++i) acc[i] = bv;
  for (int k0 = 0; k0 < Ksrc; k0 += 32) {
    bf16x16 a = load_a_frag(src, ld_src, m0, k0, lane);
    bf16x16 b = load_b_frag(wt, Ksrc, 0, k0, lane);
    acc = __builtin_amdgcn_wmma_f32_16x16x32_bf16(false, a, false, b,
                                                  (short)0, acc, false, false);
  }
  if (n < KOUT) {
#pragma unroll
    for (int r = 0; r < 8; ++r) {
      const int m = m0 + r + ((lane & 16) ? 8 : 0);
      out[(long)m * KOUT + n] = acc[r];
    }
  }
}

// Fused 4-layer MLP: [M,KIN_PAD]bf16 -> 128 relu -> 128 relu -> 128 relu -> KOUT f32.
// 128 threads = 4 wave32, 64 rows per block; each wave owns a private 16-row stripe,
// so no inter-layer barriers are needed after the cooperative input stage.
template <int KIN_PAD, int KOUT>
__global__ __launch_bounds__(128) void mlp4_kernel(
    const __bf16* __restrict__ in,
    const __bf16* __restrict__ w0t, const float* __restrict__ b0,
    const __bf16* __restrict__ w1t, const float* __restrict__ b1,
    const __bf16* __restrict__ w2t, const float* __restrict__ b2,
    const __bf16* __restrict__ w3t, const float* __restrict__ b3,
    float* __restrict__ out) {
  constexpr int H   = 128;
  constexpr int LDH = H + 8;  // 16B row skew: A-frag rows hit distinct LDS banks
  __shared__ __align__(16) __bf16 s_in[64 * KIN_PAD];
  __shared__ __align__(16) __bf16 s_a[64 * LDH];
  __shared__ __align__(16) __bf16 s_b[64 * LDH];

  const int tid  = threadIdx.x;
  const int lane = tid & 31;
  const int m0   = (tid >> 5) * 16;
  const long row0 = (long)blockIdx.x * 64;

  if (tid == 0) {  // pull the big weight matrices toward L2 (global_prefetch_b8)
    __builtin_prefetch(w1t, 0, 1);
    __builtin_prefetch(w2t, 0, 1);
  }

  // stage the input tile into LDS
#if __has_builtin(__builtin_amdgcn_global_load_async_to_lds_b128)
  {  // CDNA5 async copy: global -> LDS without touching VGPRs (ASYNCcnt-tracked)
    const int n16 = 64 * KIN_PAD / 8;  // 16B chunks
    const char* gbase = (const char*)(in + row0 * KIN_PAD);
    char* lbase = (char*)s_in;
    for (int i = tid; i < n16; i += 128) {
      __builtin_amdgcn_global_load_async_to_lds_b128(
          (AS1 i32x4*)(gbase + (size_t)i * 16),
          (AS3 i32x4*)(lbase + i * 16),
          0, 0);
    }
    wait_async0();
  }
#else
  {
    const uint4* g = (const uint4*)(in + row0 * KIN_PAD);
    uint4* l = (uint4*)s_in;
    const int n16 = 64 * KIN_PAD / 8;
    for (int i = tid; i < n16; i += 128) l[i] = g[i];
  }
#endif
  __syncthreads();

  mlp_layer(s_in, KIN_PAD, KIN_PAD, w0t, b0, s_a, LDH, H, true, m0, lane);
  mlp_layer(s_a, LDH, H, w1t, b1, s_b, LDH, H, true, m0, lane);
  mlp_layer(s_b, LDH, H, w2t, b2, s_a, LDH, H, true, m0, lane);
  mlp_layer_out<KOUT>(s_a, LDH, H, w3t, b3, out + row0 * KOUT, m0, lane);
}

// ---- transpose + pad weights: f32 [Kin][Kout] -> bf16 [NoutPad][KinPad] ----
__global__ void prep_wt_kernel(const float* __restrict__ W, __bf16* __restrict__ Wt,
                               int Kin, int Kout, int KinPad, int NoutPad) {
  const int idx = blockIdx.x * blockDim.x + threadIdx.x;
  if (idx >= KinPad * NoutPad) return;
  const int n = idx / KinPad, k = idx % KinPad;
  const float v = (n < Kout && k < Kin) ? W[k * Kout + n] : 0.0f;
  Wt[idx] = f2bf(v);
}

// ---- per-cloud kNN: cloud coords in LDS, register top-20 per target ----
__global__ __launch_bounds__(256) void knn_kernel(const float* __restrict__ h,
                                                  int* __restrict__ srcI,
                                                  float* __restrict__ dist) {
  __shared__ float sx[P_CLOUD], sy[P_CLOUD], sz[P_CLOUD];
  const int b  = blockIdx.x >> 2;                       // cloud (4 blocks / cloud)
  const int tq = ((blockIdx.x & 3) << 8) + threadIdx.x; // target point in cloud
  const float* hc = h + (long)b * P_CLOUD * 3;
  for (int j = threadIdx.x; j < P_CLOUD; j += 256) {
    sx[j] = hc[j * 3 + 0]; sy[j] = hc[j * 3 + 1]; sz[j] = hc[j * 3 + 2];
  }
  __syncthreads();
  const float px = sx[tq], py = sy[tq], pz = sz[tq];
  float bd[KNN]; int bi[KNN];
#pragma unroll
  for (int i = 0; i < KNN; ++i) { bd[i] = 3.0e38f; bi[i] = 0; }
  for (int j = 0; j < P_CLOUD; ++j) {
    const float dx = sx[j] - px, dy = sy[j] - py, dz = sz[j] - pz;
    float d2 = dx * dx + dy * dy + dz * dz;
    if (j == tq) d2 = __builtin_inff();                 // exclude self-loop
    if (d2 < bd[KNN - 1]) {                             // branchless bubble insert
      float cd = d2; int ci = j;
#pragma unroll
      for (int i = 0; i < KNN; ++i) {
        const bool sw = cd < bd[i];
        const float td = sw ? bd[i] : cd;
        const int   ti = sw ? bi[i] : ci;
        bd[i] = sw ? cd : bd[i];
        bi[i] = sw ? ci : bi[i];
        cd = td; ci = ti;
      }
    }
  }
  const long e0 = ((long)b * P_CLOUD + tq) * KNN;
#pragma unroll
  for (int i = 0; i < KNN; ++i) {
    srcI[e0 + i] = b * P_CLOUD + bi[i];
    dist[e0 + i] = sqrtf(bd[i]);
  }
}

// ---- edge/node input assembly (bf16, zero-padded to K=32) ----
__global__ void build_edge_in0(const float* __restrict__ h, const int* __restrict__ srcI,
                               const float* __restrict__ dist, __bf16* __restrict__ e_in, int E) {
  const int e = blockIdx.x * blockDim.x + threadIdx.x;
  if (e >= E) return;
  const int col = e / KNN;
  const int row = srcI[e];
  __bf16* o = e_in + (long)e * 32;
  o[0] = f2bf(h[row * 3 + 0]); o[1] = f2bf(h[row * 3 + 1]); o[2] = f2bf(h[row * 3 + 2]);
  o[3] = f2bf(h[col * 3 + 0]); o[4] = f2bf(h[col * 3 + 1]); o[5] = f2bf(h[col * 3 + 2]);
  o[6] = f2bf(dist[e]);
  const __bf16 z = f2bf(0.0f);
#pragma unroll
  for (int i = 7; i < 32; ++i) o[i] = z;
}

__global__ void build_node_in0(const float* __restrict__ h, const float* __restrict__ eemb,
                               __bf16* __restrict__ nin, int N) {
  const int t = blockIdx.x * blockDim.x + threadIdx.x;
  if (t >= N * 16) return;
  const int i = t >> 4, f = t & 15;
  const float* p = eemb + ((long)i * KNN) * 16 + f;   // edges target-major -> direct segsum
  float s = 0.0f;
#pragma unroll
  for (int jj = 0; jj < KNN; ++jj) s += p[jj * 16];
  __bf16* o = nin + (long)i * 32;
  o[3 + f] = f2bf(s);
  if (f < 3)  o[f] = f2bf(h[i * 3 + f]);
  if (f < 13) o[19 + f] = f2bf(0.0f);
}

__global__ void build_edge_in1(const float* __restrict__ h1, const int* __restrict__ srcI,
                               const float* __restrict__ eemb, __bf16* __restrict__ e_in, int E) {
  const int e = blockIdx.x * blockDim.x + threadIdx.x;
  if (e >= E) return;
  const int col = e / KNN;
  const int row = srcI[e];
  __bf16* o = e_in + (long)e * 32;
  o[0] = f2bf(h1[row]);
  o[1] = f2bf(h1[col]);
  const float* em = eemb + (long)e * 16;
#pragma unroll
  for (int f = 0; f < 16; ++f) o[2 + f] = f2bf(em[f]);
  const __bf16 z = f2bf(0.0f);
#pragma unroll
  for (int i = 18; i < 32; ++i) o[i] = z;
}

__global__ void build_node_in1(const float* __restrict__ h1, const float* __restrict__ eemb,
                               __bf16* __restrict__ nin, int N) {
  const int t = blockIdx.x * blockDim.x + threadIdx.x;
  if (t >= N * 16) return;
  const int i = t >> 4, f = t & 15;
  const float* p = eemb + ((long)i * KNN) * 16 + f;
  float s = 0.0f;
#pragma unroll
  for (int jj = 0; jj < KNN; ++jj) s += p[jj * 16];
  __bf16* o = nin + (long)i * 32;
  o[1 + f] = f2bf(s);
  if (f == 0) o[0] = f2bf(h1[i]);
  if (f < 15) o[17 + f] = f2bf(0.0f);
}

// ---- per-cloud mean pool + 1x1 readout ----
__global__ __launch_bounds__(256) void pool_readout(const float* __restrict__ h2,
                                                    const float* __restrict__ Wr,
                                                    const float* __restrict__ br,
                                                    float* __restrict__ out) {
  __shared__ float red[256];
  const int b = blockIdx.x, tid = threadIdx.x;
  float s = 0.0f;
  for (int j = tid; j < P_CLOUD; j += 256) s += h2[b * P_CLOUD + j];
  red[tid] = s;
  __syncthreads();
  for (int w = 128; w > 0; w >>= 1) {
    if (tid < w) red[tid] += red[tid + w];
    __syncthreads();
  }
  if (tid == 0) out[b] = (red[0] * (1.0f / P_CLOUD)) * Wr[0] + br[0];
}

extern "C" void kernel_launch(void* const* d_in, const int* in_sizes, int n_in,
                              void* d_out, int out_size, void* d_ws, size_t ws_size,
                              hipStream_t stream) {
  (void)in_sizes; (void)n_in; (void)out_size; (void)ws_size;
  const int B = NCLOUD, P = P_CLOUD, N = B * P, E = N * KNN;
  const float* h = (const float*)d_in[0];

  // params flattened in pytree order: layers[0].edge(W,b)x4, layers[0].node x4,
  // layers[1].edge x4, layers[1].node x4, readout (W,b)
  const float* W[16]; const float* bias[16];
  for (int i = 0; i < 16; ++i) {
    W[i]    = (const float*)d_in[3 + 2 * i];
    bias[i] = (const float*)d_in[4 + 2 * i];
  }
  const float* Wr = (const float*)d_in[35];
  const float* br = (const float*)d_in[36];

  char* base = (char*)d_ws;
  size_t off = 0;
  auto alloc = [&](size_t bytes) -> void* {
    off = (off + 255) & ~(size_t)255;
    void* p = base + off;
    off += bytes;
    return p;
  };

  int*    srcI = (int*)   alloc((size_t)E * 4);
  float*  dist = (float*) alloc((size_t)E * 4);
  __bf16* e_in = (__bf16*)alloc((size_t)E * 32 * 2);
  float*  eemb = (float*) alloc((size_t)E * 16 * 4);
  __bf16* nin  = (__bf16*)alloc((size_t)N * 32 * 2);
  float*  h1   = (float*) alloc((size_t)N * 4);
  float*  h2   = (float*) alloc((size_t)N * 4);

  __bf16* wt[16];
  for (int m = 0; m < 4; ++m) {
    wt[4 * m + 0] = (__bf16*)alloc(128 * 32 * 2);
    wt[4 * m + 1] = (__bf16*)alloc(128 * 128 * 2);
    wt[4 * m + 2] = (__bf16*)alloc(128 * 128 * 2);
    wt[4 * m + 3] = (__bf16*)alloc(16 * 128 * 2);
  }

  auto prep = [&](const float* Wp, __bf16* Wt, int Kin, int Kout, int KinPad, int NoutPad) {
    const int total = KinPad * NoutPad;
    prep_wt_kernel<<<(total + 255) / 256, 256, 0, stream>>>(Wp, Wt, Kin, Kout, KinPad, NoutPad);
  };
  const int kin0[4] = {7, 19, 18, 17};  // first-layer Kin: L0.edge, L0.node, L1.edge, L1.node
  for (int m = 0; m < 4; ++m) {
    const int kout3 = (m == 0 || m == 2) ? 16 : 1;
    prep(W[4 * m + 0], wt[4 * m + 0], kin0[m], 128, 32, 128);
    prep(W[4 * m + 1], wt[4 * m + 1], 128, 128, 128, 128);
    prep(W[4 * m + 2], wt[4 * m + 2], 128, 128, 128, 128);
    prep(W[4 * m + 3], wt[4 * m + 3], 128, kout3, 128, 16);
  }

  knn_kernel<<<N / 256, 256, 0, stream>>>(h, srcI, dist);

  // ---- layer 0 ----
  build_edge_in0<<<(E + 255) / 256, 256, 0, stream>>>(h, srcI, dist, e_in, E);
  mlp4_kernel<32, 16><<<E / 64, 128, 0, stream>>>(e_in, wt[0], bias[0], wt[1], bias[1],
                                                  wt[2], bias[2], wt[3], bias[3], eemb);
  build_node_in0<<<(N * 16 + 255) / 256, 256, 0, stream>>>(h, eemb, nin, N);
  mlp4_kernel<32, 1><<<N / 64, 128, 0, stream>>>(nin, wt[4], bias[4], wt[5], bias[5],
                                                 wt[6], bias[6], wt[7], bias[7], h1);
  // ---- layer 1 ----
  build_edge_in1<<<(E + 255) / 256, 256, 0, stream>>>(h1, srcI, eemb, e_in, E);
  mlp4_kernel<32, 16><<<E / 64, 128, 0, stream>>>(e_in, wt[8], bias[8], wt[9], bias[9],
                                                  wt[10], bias[10], wt[11], bias[11], eemb);
  build_node_in1<<<(N * 16 + 255) / 256, 256, 0, stream>>>(h1, eemb, nin, N);
  mlp4_kernel<32, 1><<<N / 64, 128, 0, stream>>>(nin, wt[12], bias[12], wt[13], bias[13],
                                                 wt[14], bias[14], wt[15], bias[15], h2);

  pool_readout<<<B, 256, 0, stream>>>(h2, Wr, br, (float*)d_out);
}
